// ContrastiveLoss_18399639896499
// MI455X (gfx1250) — matrix operand order repaired and checked by reading, not compiled, not run on hardware
//
#include <hip/hip_runtime.h>

typedef __attribute__((ext_vector_type(16))) _Float16 v16h;
typedef __attribute__((ext_vector_type(8)))  _Float16 v8h;
typedef __attribute__((ext_vector_type(8)))  float    v8f;

#define NROW 8192
#define DDIM 256

__device__ __forceinline__ float wave_reduce_add32(float v) {
  v += __shfl_xor(v, 1, 32);
  v += __shfl_xor(v, 2, 32);
  v += __shfl_xor(v, 4, 32);
  v += __shfl_xor(v, 8, 32);
  v += __shfl_xor(v, 16, 32);
  return v;
}

// ---------------------------------------------------------------------------
// Kernel 1: L2-normalize rows of anc|pos|neg (fp32) -> row-major _Float16.
// ---------------------------------------------------------------------------
__global__ __launch_bounds__(256) void norm_f16_kernel(
    const float* __restrict__ anc, const float* __restrict__ pos,
    const float* __restrict__ neg, _Float16* __restrict__ dst16) {
  const int wave = threadIdx.x >> 5;
  const int lane = threadIdx.x & 31;
  const int row  = blockIdx.x * 8 + wave;            // 0 .. 3*NROW-1
  const float* src;
  if (row < NROW)            src = anc + (size_t)row * DDIM;
  else if (row < 2 * NROW)   src = pos + (size_t)(row - NROW) * DDIM;
  else                       src = neg + (size_t)(row - 2 * NROW) * DDIM;
  const float4* s4 = (const float4*)src;
  float4 x0 = s4[lane * 2];
  float4 x1 = s4[lane * 2 + 1];
  float ss = x0.x*x0.x + x0.y*x0.y + x0.z*x0.z + x0.w*x0.w
           + x1.x*x1.x + x1.y*x1.y + x1.z*x1.z + x1.w*x1.w;
  ss = wave_reduce_add32(ss);
  const float sc = 1.0f / fmaxf(sqrtf(ss), 1e-8f);   // torch eps semantics
  v8h h;
  h[0] = (_Float16)(x0.x * sc); h[1] = (_Float16)(x0.y * sc);
  h[2] = (_Float16)(x0.z * sc); h[3] = (_Float16)(x0.w * sc);
  h[4] = (_Float16)(x1.x * sc); h[5] = (_Float16)(x1.y * sc);
  h[6] = (_Float16)(x1.z * sc); h[7] = (_Float16)(x1.w * sc);
  ((v8h*)(dst16 + (size_t)row * DDIM))[lane] = h;
}

// ---------------------------------------------------------------------------
// Kernel 2: exact fp32 diagonal logit 20*cos(a_i,p_i); also zero S[i].
// ---------------------------------------------------------------------------
__global__ __launch_bounds__(256) void diag_kernel(
    const float* __restrict__ anc, const float* __restrict__ pos,
    float* __restrict__ S, float* __restrict__ diagv) {
  const int wave = threadIdx.x >> 5;
  const int lane = threadIdx.x & 31;
  const int row  = blockIdx.x * 8 + wave;            // 0 .. NROW-1
  const float4* a4 = (const float4*)(anc + (size_t)row * DDIM);
  const float4* p4 = (const float4*)(pos + (size_t)row * DDIM);
  float4 a0 = a4[lane * 2], a1 = a4[lane * 2 + 1];
  float4 p0 = p4[lane * 2], p1 = p4[lane * 2 + 1];
  float aa = a0.x*a0.x + a0.y*a0.y + a0.z*a0.z + a0.w*a0.w
           + a1.x*a1.x + a1.y*a1.y + a1.z*a1.z + a1.w*a1.w;
  float pp = p0.x*p0.x + p0.y*p0.y + p0.z*p0.z + p0.w*p0.w
           + p1.x*p1.x + p1.y*p1.y + p1.z*p1.z + p1.w*p1.w;
  float ap = a0.x*p0.x + a0.y*p0.y + a0.z*p0.z + a0.w*p0.w
           + a1.x*p1.x + a1.y*p1.y + a1.z*p1.z + a1.w*p1.w;
  aa = wave_reduce_add32(aa);
  pp = wave_reduce_add32(pp);
  ap = wave_reduce_add32(ap);
  if (lane == 0) {
    diagv[row] = 20.0f * ap / (fmaxf(sqrtf(aa), 1e-8f) * fmaxf(sqrtf(pp), 1e-8f));
    S[row] = 0.0f;
  }
}

// ---------------------------------------------------------------------------
// Kernel 3: fused GEMM + fixed-shift softmax partial sums.
// Block = 8 waves x 32 rows = 256 rows. Each wave keeps TWO register-resident
// 16-row A tiles (full K=256). B tiles (16 cols x 256 K = 8 KB) are staged
// into double-buffered LDS with GLOBAL_LOAD_ASYNC_TO_LDS_B128 (one copy per
// block instead of one per wave), padded to a 528-byte row stride so the
// b128 fragment reads are LDS-bank-conflict-free. Prefetch of tile ct+1
// overlaps 16 WMMAs on tile ct; one s_wait_asynccnt + barrier per tile.
// ---------------------------------------------------------------------------
#define LDS_ROW_H 264              // 256 halves + 8 halves pad (528 B stride)
#define LDS_BUF_B (16 * LDS_ROW_H * 2)   // 8448 bytes per buffer

__global__ __launch_bounds__(256) void gemm_lse_kernel(
    const _Float16* __restrict__ a16, const _Float16* __restrict__ p16,
    const _Float16* __restrict__ n16, float* __restrict__ S) {
  __shared__ __align__(16) _Float16 bt[2][16][LDS_ROW_H];

  const int wave = threadIdx.x >> 5;
  const int lane = threadIdx.x & 31;
  const int lm   = lane & 15;
  const int hl   = lane >> 4;
  const int r0   = blockIdx.x * 256 + wave * 32;     // two row tiles: r0, r0+16

  // Register-resident A fragments for both row tiles, all 8 K-chunks.
  // ISA 16-bit A 16x32 layout: lanes 0-15 K{kb+0..7, kb+16..23},
  // lanes 16-31 K{kb+8..15, kb+24..31}.
  v16h Af[2][8];
  #pragma unroll
  for (int t = 0; t < 2; ++t) {
    const _Float16* abase = a16 + (size_t)(r0 + t * 16 + lm) * DDIM + hl * 8;
    #pragma unroll
    for (int kc = 0; kc < 8; ++kc) {
      v8h lo = *(const v8h*)(abase + kc * 32);
      v8h hi = *(const v8h*)(abase + kc * 32 + 16);
      Af[t][kc] = __builtin_shufflevector(lo, hi,
                      0,1,2,3,4,5,6,7,8,9,10,11,12,13,14,15);
    }
  }

  float s0[8], s1[8];
  #pragma unroll
  for (int r = 0; r < 8; ++r) { s0[r] = 0.0f; s1[r] = 0.0f; }

  // Low 32 bits of the flat shared-pointer == LDS offset (aperture mapping).
  const unsigned ldsbase = (unsigned)(unsigned long long)(void*)&bt[0][0][0];
  // Wave w stages rows w and w+8 of the tile; 16B per lane per row.
  const unsigned lds_wave = ldsbase + (unsigned)(wave * (LDS_ROW_H * 2) + lane * 16);

  const int ct0   = blockIdx.y * 128;                // 1024 column tiles total
  const int ctend = ct0 + 128;

  auto issue_tile = [&](int ct, int buf) {
    const _Float16* Bm = (ct < 512) ? p16 : n16;
    const _Float16* g  = Bm + (size_t)((ct & 511) * 16 + wave) * DDIM + lane * 8;
    unsigned l0 = lds_wave + (unsigned)(buf * LDS_BUF_B);
    unsigned long long g0 = (unsigned long long)(size_t)g;
    unsigned long long g1 = (unsigned long long)(size_t)(g + 8 * DDIM);
    asm volatile("global_load_async_to_lds_b128 %0, %1, off"
                 :: "v"(l0), "v"(g0) : "memory");
    asm volatile("global_load_async_to_lds_b128 %0, %1, off"
                 :: "v"(l0 + (unsigned)(8 * LDS_ROW_H * 2)), "v"(g1) : "memory");
  };

  // Prologue: stage the first tile.
  issue_tile(ct0, 0);
  asm volatile("s_wait_asynccnt 0x0" ::: "memory");
  __syncthreads();

  const float kL2E = 28.853900817779268f;            // (1/TEMP)*log2(e)

  for (int ct = ct0; ct < ctend; ++ct) {
    const int cur = ct & 1;
    if (ct + 1 < ctend) issue_tile(ct + 1, cur ^ 1);

    v8f acc0 = {}, acc1 = {};
    #pragma unroll
    for (int kc = 0; kc < 8; ++kc) {
      // B 32x16 layout: lanes 0-15 K{kb+0..7, kb+16..23} of column lm; lanes 16-31 K{kb+8..15, kb+24..31}.
      const v8h* bp = (const v8h*)&bt[cur][lm][hl * 8 + kc * 32];
      v16h b = __builtin_shufflevector(bp[0], bp[2],
                   0,1,2,3,4,5,6,7,8,9,10,11,12,13,14,15);
      acc0 = __builtin_amdgcn_wmma_f32_16x16x32_f16(
          false, Af[0][kc], false, b, (short)0, acc0, false, false);
      acc1 = __builtin_amdgcn_wmma_f32_16x16x32_f16(
          false, Af[1][kc], false, b, (short)0, acc1, false, false);
    }
    #pragma unroll
    for (int r = 0; r < 8; ++r) {
      s0[r] += __builtin_amdgcn_exp2f(__builtin_fmaf(acc0[r], kL2E, -kL2E));
      s1[r] += __builtin_amdgcn_exp2f(__builtin_fmaf(acc1[r], kL2E, -kL2E));
    }

    // My prefetch done; barrier => everyone's prefetch done and everyone is
    // finished reading the buffer we overwrite next iteration.
    asm volatile("s_wait_asynccnt 0x0" ::: "memory");
    __syncthreads();
  }

  // C layout: VGPR r, lanes 0-15 = row r, lanes 16-31 = row r+8.
  #pragma unroll
  for (int r = 0; r < 8; ++r) {
    float v0 = s0[r], v1 = s1[r];
    v0 += __shfl_xor(v0, 1, 32); v1 += __shfl_xor(v1, 1, 32);
    v0 += __shfl_xor(v0, 2, 32); v1 += __shfl_xor(v1, 2, 32);
    v0 += __shfl_xor(v0, 4, 32); v1 += __shfl_xor(v1, 4, 32);
    v0 += __shfl_xor(v0, 8, 32); v1 += __shfl_xor(v1, 8, 32);
    if (lm == 0) {
      atomicAdd(&S[r0 + hl * 8 + r], v0);
      atomicAdd(&S[r0 + 16 + hl * 8 + r], v1);
    }
  }
}

// ---------------------------------------------------------------------------
// Kernel 4: loss = mean_i( 20 + log(S_i) - diag_i )
// ---------------------------------------------------------------------------
__global__ __launch_bounds__(256) void finalize_kernel(
    const float* __restrict__ S, const float* __restrict__ diagv,
    float* __restrict__ out) {
  __shared__ float red[256];
  float acc = 0.0f;
  for (int i = threadIdx.x; i < NROW; i += 256)
    acc += 20.0f + logf(S[i]) - diagv[i];
  red[threadIdx.x] = acc;
  __syncthreads();
  for (int t = 128; t > 0; t >>= 1) {
    if (threadIdx.x < t) red[threadIdx.x] += red[threadIdx.x + t];
    __syncthreads();
  }
  if (threadIdx.x == 0) out[0] = red[0] / (float)NROW;
}

extern "C" void kernel_launch(void* const* d_in, const int* in_sizes, int n_in,
                              void* d_out, int out_size, void* d_ws, size_t ws_size,
                              hipStream_t stream) {
  const float* anc = (const float*)d_in[0];
  const float* pos = (const float*)d_in[1];
  const float* neg = (const float*)d_in[2];
  float* out = (float*)d_out;

  char* ws = (char*)d_ws;
  _Float16* f16all = (_Float16*)ws;                       // 3*8192*256 halves = 12 MiB
  _Float16* a16 = f16all;
  _Float16* p16 = f16all + (size_t)NROW * DDIM;
  _Float16* n16 = f16all + (size_t)2 * NROW * DDIM;
  float* S    = (float*)(ws + (size_t)3 * NROW * DDIM * sizeof(_Float16));
  float* diag = S + NROW;

  norm_f16_kernel<<<3 * NROW / 8, 256, 0, stream>>>(anc, pos, neg, f16all);
  diag_kernel<<<NROW / 8, 256, 0, stream>>>(anc, pos, S, diag);
  gemm_lse_kernel<<<dim3(NROW / 256, 8), 256, 0, stream>>>(a16, p16, n16, S);
  finalize_kernel<<<1, 256, 0, stream>>>(S, diag, out);
}
